// RelativeMultiHeadAttention_21749714387556
// MI455X (gfx1250) — matrix operand order repaired and checked by reading, not compiled, not run on hardware
//
#include <hip/hip_runtime.h>

#define DM   1024
#define NH   16
#define DK   64
#define SEQ  1000
#define BATCH 4
#define NROWS (BATCH*SEQ)     // 4000
#define NIB  63               // ceil(SEQ/16)
#define WROWS 1999            // 2S-1 window rows

typedef __attribute__((ext_vector_type(16))) _Float16 v16h;
typedef __attribute__((ext_vector_type(8)))  float    v8f;

// ---- CDNA5 WMMA fragment loaders (16x16x32 f16, wave32) -------------------
// A (16x32): lane = row (lane&15); lanes<16 hold K {0..7,16..23},
// lanes>=16 hold K {8..15,24..31}  -> two contiguous 16B chunks.
__device__ __forceinline__ v16h load_a_frag(const _Float16* base, int lane) {
  const int lo = (lane < 16) ? 0 : 8;
  union { uint4 u[2]; v16h h; } f;
  f.u[0] = *reinterpret_cast<const uint4*>(base + lo);
  f.u[1] = *reinterpret_cast<const uint4*>(base + lo + 16);
  return f.h;
}
// B (32x16): lane = col (lane&15); lanes<16 hold K {0..7,16..23},
// lanes>=16 hold K {8..15,24..31} (same K order as A).
__device__ __forceinline__ v16h load_b_frag(const _Float16* base, int lane) {
  const int off = (lane < 16) ? 0 : 8;
  union { uint4 u[2]; v16h h; } f;
  f.u[0] = *reinterpret_cast<const uint4*>(base + off);
  f.u[1] = *reinterpret_cast<const uint4*>(base + off + 16);
  return f.h;
}

// ---- f32 -> f16 convert ----------------------------------------------------
__global__ void cvt_f16_kernel(const float* __restrict__ src,
                               _Float16* __restrict__ dst, int n) {
  int i = blockIdx.x * blockDim.x + threadIdx.x;
  if (i < n) dst[i] = (_Float16)src[i];
}

// ---- LayerNorm (fp32 math) + cast to f16 ----------------------------------
__global__ __launch_bounds__(256) void ln_f16_kernel(
    const float* __restrict__ x, const float* __restrict__ g,
    const float* __restrict__ bta, _Float16* __restrict__ y) {
  const int row = blockIdx.x;
  const int t = threadIdx.x;                   // 256 threads, 4 elems each
  float4 v = reinterpret_cast<const float4*>(x + row * DM)[t];
  float s  = v.x + v.y + v.z + v.w;
  float ss = v.x*v.x + v.y*v.y + v.z*v.z + v.w*v.w;
  for (int o = 1; o < 32; o <<= 1) { s += __shfl_xor(s, o, 32); ss += __shfl_xor(ss, o, 32); }
  __shared__ float sb[8], qb[8];
  if ((t & 31) == 0) { sb[t >> 5] = s; qb[t >> 5] = ss; }
  __syncthreads();
  float tot = 0.f, tot2 = 0.f;
  #pragma unroll
  for (int i = 0; i < 8; ++i) { tot += sb[i]; tot2 += qb[i]; }
  const float mu = tot * (1.f / DM);
  const float var = tot2 * (1.f / DM) - mu * mu;
  const float rs = rsqrtf(var + 1e-5f);
  float4 gg = reinterpret_cast<const float4*>(g)[t];
  float4 bb = reinterpret_cast<const float4*>(bta)[t];
  _Float16* yr = y + row * DM + t * 4;
  yr[0] = (_Float16)((v.x - mu) * rs * gg.x + bb.x);
  yr[1] = (_Float16)((v.y - mu) * rs * gg.y + bb.y);
  yr[2] = (_Float16)((v.z - mu) * rs * gg.z + bb.z);
  yr[3] = (_Float16)((v.w - mu) * rs * gg.w + bb.w);
}

// ---- GEMM  Y[m,n] = sum_k A[m,k]*W[n,k] + bias[n]  (NT, K=1024) ------------
// Per-wave output tile: 32x64 (2 M-tiles x 4 N-tiles, 8 accumulators).
// Per K-step: 2 A-frags + 4 B-frags (6KB) feed 8 WMMAs -> 0.75 KB/wmma.
// mode 0: write f16 Y; mode 1: scatter f16 into V^T [B,H,64,1024];
// mode 2: write f32 Y + residual (final projection).
__global__ __launch_bounds__(128) void gemm_wmma_kernel(
    const _Float16* __restrict__ A, const _Float16* __restrict__ W,
    const float* __restrict__ bias, const float* __restrict__ res,
    _Float16* __restrict__ outH, _Float16* __restrict__ vT,
    float* __restrict__ outF, int mode) {
  const int lane = threadIdx.x & 31;
  const int wave = threadIdx.x >> 5;
  const int m0 = blockIdx.x * 32;                    // 32-row strip
  const int n0 = (blockIdx.y * 4 + wave) * 64;       // 64-col strip per wave
  const int col = lane & 15;

  const _Float16* arow0 = A + (m0      + col) * DM;
  const _Float16* arow1 = A + (m0 + 16 + col) * DM;
  const _Float16* brow0 = W + (n0      + col) * DM;
  const _Float16* brow1 = W + (n0 + 16 + col) * DM;
  const _Float16* brow2 = W + (n0 + 32 + col) * DM;
  const _Float16* brow3 = W + (n0 + 48 + col) * DM;

  v8f acc[2][4];
  #pragma unroll
  for (int mi = 0; mi < 2; ++mi)
    #pragma unroll
    for (int nt = 0; nt < 4; ++nt) acc[mi][nt] = (v8f){};

  for (int k0 = 0; k0 < DM; k0 += 32) {
    // prefetch one K-panel (512 halves) ahead
    __builtin_prefetch(arow0 + k0 + 512, 0, 1);
    __builtin_prefetch(brow0 + k0 + 512, 0, 1);
    const v16h a0 = load_a_frag(arow0 + k0, lane);
    const v16h a1 = load_a_frag(arow1 + k0, lane);
    const v16h b0 = load_b_frag(brow0 + k0, lane);
    const v16h b1 = load_b_frag(brow1 + k0, lane);
    const v16h b2 = load_b_frag(brow2 + k0, lane);
    const v16h b3 = load_b_frag(brow3 + k0, lane);
    acc[0][0] = __builtin_amdgcn_wmma_f32_16x16x32_f16(false, a0, false, b0, (short)0, acc[0][0], false, false);
    acc[0][1] = __builtin_amdgcn_wmma_f32_16x16x32_f16(false, a0, false, b1, (short)0, acc[0][1], false, false);
    acc[0][2] = __builtin_amdgcn_wmma_f32_16x16x32_f16(false, a0, false, b2, (short)0, acc[0][2], false, false);
    acc[0][3] = __builtin_amdgcn_wmma_f32_16x16x32_f16(false, a0, false, b3, (short)0, acc[0][3], false, false);
    acc[1][0] = __builtin_amdgcn_wmma_f32_16x16x32_f16(false, a1, false, b0, (short)0, acc[1][0], false, false);
    acc[1][1] = __builtin_amdgcn_wmma_f32_16x16x32_f16(false, a1, false, b1, (short)0, acc[1][1], false, false);
    acc[1][2] = __builtin_amdgcn_wmma_f32_16x16x32_f16(false, a1, false, b2, (short)0, acc[1][2], false, false);
    acc[1][3] = __builtin_amdgcn_wmma_f32_16x16x32_f16(false, a1, false, b3, (short)0, acc[1][3], false, false);
  }

  const int rowoff = (lane < 16) ? 0 : 8;
  #pragma unroll
  for (int mi = 0; mi < 2; ++mi) {
    #pragma unroll
    for (int nt = 0; nt < 4; ++nt) {
      const int nn = n0 + nt * 16 + col;
      const float bv = bias[nn];
      #pragma unroll
      for (int r = 0; r < 8; ++r) {
        const int m = m0 + mi * 16 + rowoff + r;
        const float val = acc[mi][nt][r] + bv;
        if (mode == 0) {
          outH[m * DM + nn] = (_Float16)val;
        } else if (mode == 1) {
          const int bI = m / SEQ, iI = m - bI * SEQ;
          const int hI = nn >> 6, dI = nn & 63;
          vT[(((bI * NH + hI) * DK + dI) << 10) + iI] = (_Float16)val;
        } else {
          outF[m * DM + nn] = val + res[m * DM + nn];
        }
      }
    }
  }
}

// ---- Fused flash attention with relative bias ------------------------------
// One wave handles one (b,h, 16-row i-block); loops j in 32-col chunks.
__global__ __launch_bounds__(32) void attn_kernel(
    const _Float16* __restrict__ Q,    // [B,S,DM]  (q proj, f16)
    const _Float16* __restrict__ K,    // [B,S,DM]  (k proj, f16)
    const _Float16* __restrict__ VT,   // [B,H,64,1024] (v proj transposed)
    const _Float16* __restrict__ WIN,  // [1999,64] f16 window
    _Float16* __restrict__ O) {        // [B,S,DM] attention output
  const int lane = threadIdx.x & 31;
  const int ib = blockIdx.x % NIB;
  const int bh = blockIdx.x / NIB;
  const int h = bh & (NH - 1), b = bh >> 4;
  const int i0 = ib * 16;
  const int rowoff = (lane < 16) ? 0 : 8;   // C-layout row offset
  const int col = lane & 15;                // C-layout column / B-frag column

  __shared__ __align__(16) float    p_lds[16][48];  // bias GEMM staging
  __shared__ __align__(16) _Float16 a_lds[16][32];  // attn-weight transpose

  // Q block fragments (K=64 -> two K32 frags), row clamped for the tail block
  int qi = i0 + col; if (qi > SEQ - 1) qi = SEQ - 1;
  const _Float16* qrow = Q + (b * SEQ + qi) * DM + h * DK;
  const v16h aq0 = load_a_frag(qrow, lane);
  const v16h aq1 = load_a_frag(qrow + 32, lane);

  float m_run[8], l_run[8];
  v8f o0 = {}, o1 = {}, o2 = {}, o3 = {};
  #pragma unroll
  for (int r = 0; r < 8; ++r) { m_run[r] = -1e30f; l_run[r] = 0.f; }
  const float scale = 0.125f;   // 1/sqrt(64)
  const int vrow0 = (b * NH + h) * DK;

  for (int jc = 0; jc < 1024; jc += 32) {
    // -- QK for two 16-col tiles
    v8f c0 = {}, c1 = {};
    {
      int j = jc + col;      if (j > SEQ - 1) j = SEQ - 1;
      const _Float16* krow = K + (b * SEQ + j) * DM + h * DK;
      c0 = __builtin_amdgcn_wmma_f32_16x16x32_f16(false, aq0, false,
            load_b_frag(krow, lane), (short)0, c0, false, false);
      c0 = __builtin_amdgcn_wmma_f32_16x16x32_f16(false, aq1, false,
            load_b_frag(krow + 32, lane), (short)0, c0, false, false);
    }
    {
      int j = jc + 16 + col; if (j > SEQ - 1) j = SEQ - 1;
      const _Float16* krow = K + (b * SEQ + j) * DM + h * DK;
      c1 = __builtin_amdgcn_wmma_f32_16x16x32_f16(false, aq0, false,
            load_b_frag(krow, lane), (short)0, c1, false, false);
      c1 = __builtin_amdgcn_wmma_f32_16x16x32_f16(false, aq1, false,
            load_b_frag(krow + 32, lane), (short)0, c1, false, false);
    }
    // -- relative-bias panel P[di,t] = q[i0+di] . win[rbase+t], t in [0,48)
    const int rbase = i0 - jc + (SEQ - 1) - 31;
    #pragma unroll
    for (int pt = 0; pt < 3; ++pt) {
      const int t = pt * 16 + col;
      int rr = rbase + t; rr = rr < 0 ? 0 : (rr > WROWS - 1 ? WROWS - 1 : rr);
      const _Float16* wrow = WIN + rr * DK;
      v8f p = {};
      p = __builtin_amdgcn_wmma_f32_16x16x32_f16(false, aq0, false,
            load_b_frag(wrow, lane), (short)0, p, false, false);
      p = __builtin_amdgcn_wmma_f32_16x16x32_f16(false, aq1, false,
            load_b_frag(wrow + 32, lane), (short)0, p, false, false);
      #pragma unroll
      for (int r = 0; r < 8; ++r) p_lds[rowoff + r][t] = p[r];
    }
    __syncthreads();
    // -- scores + online softmax (row stats live in the 16-lane col group)
    #pragma unroll
    for (int r = 0; r < 8; ++r) {
      const int di = rowoff + r;
      float s0 = (c0[r] + p_lds[di][31 + di - col]) * scale;
      float s1 = (c1[r] + p_lds[di][31 + di - (16 + col)]) * scale;
      if (jc + col      >= SEQ) s0 = -1e30f;
      if (jc + 16 + col >= SEQ) s1 = -1e30f;
      float mx = fmaxf(s0, s1);
      for (int o = 1; o < 16; o <<= 1) mx = fmaxf(mx, __shfl_xor(mx, o, 32));
      const float mnew = fmaxf(m_run[r], mx);
      const float a0 = __expf(s0 - mnew), a1 = __expf(s1 - mnew);
      float sm = a0 + a1;
      for (int o = 1; o < 16; o <<= 1) sm += __shfl_xor(sm, o, 32);
      const float alpha = __expf(m_run[r] - mnew);
      l_run[r] = l_run[r] * alpha + sm;
      m_run[r] = mnew;
      o0[r] *= alpha; o1[r] *= alpha; o2[r] *= alpha; o3[r] *= alpha;
      a_lds[di][col] = (_Float16)a0;
      a_lds[di][16 + col] = (_Float16)a1;
    }
    __syncthreads();
    // -- PV: attn tile (C-layout) re-read from LDS in A-layout, K=32 = j-chunk
    const v16h aat = load_a_frag(&a_lds[col][0], lane);
    o0 = __builtin_amdgcn_wmma_f32_16x16x32_f16(false, aat, false,
          load_b_frag(VT + (vrow0 +  0 + col) * 1024 + jc, lane), (short)0, o0, false, false);
    o1 = __builtin_amdgcn_wmma_f32_16x16x32_f16(false, aat, false,
          load_b_frag(VT + (vrow0 + 16 + col) * 1024 + jc, lane), (short)0, o1, false, false);
    o2 = __builtin_amdgcn_wmma_f32_16x16x32_f16(false, aat, false,
          load_b_frag(VT + (vrow0 + 32 + col) * 1024 + jc, lane), (short)0, o2, false, false);
    o3 = __builtin_amdgcn_wmma_f32_16x16x32_f16(false, aat, false,
          load_b_frag(VT + (vrow0 + 48 + col) * 1024 + jc, lane), (short)0, o3, false, false);
  }
  // -- normalize + store (f16, head-interleaved back to [b,i,h*64+d])
  #pragma unroll
  for (int r = 0; r < 8; ++r) {
    const int i = i0 + rowoff + r;
    if (i < SEQ) {
      const float inv = 1.f / l_run[r];
      _Float16* orow = O + (b * SEQ + i) * DM + h * DK;
      orow[ 0 + col] = (_Float16)(o0[r] * inv);
      orow[16 + col] = (_Float16)(o1[r] * inv);
      orow[32 + col] = (_Float16)(o2[r] * inv);
      orow[48 + col] = (_Float16)(o3[r] * inv);
    }
  }
}

// ---------------------------------------------------------------------------
extern "C" void kernel_launch(void* const* d_in, const int* in_sizes, int n_in,
                              void* d_out, int out_size, void* d_ws, size_t ws_size,
                              hipStream_t stream) {
  const float* q    = (const float*)d_in[0];
  const float* k    = (const float*)d_in[1];
  const float* v    = (const float*)d_in[2];
  const float* ln_g = (const float*)d_in[3];
  const float* ln_b = (const float*)d_in[4];
  const float* wq   = (const float*)d_in[5];
  const float* bq   = (const float*)d_in[6];
  const float* wk   = (const float*)d_in[7];
  const float* bk   = (const float*)d_in[8];
  const float* wv   = (const float*)d_in[9];
  const float* bv   = (const float*)d_in[10];
  const float* wo   = (const float*)d_in[11];
  const float* bo   = (const float*)d_in[12];
  const float* rel  = (const float*)d_in[13];
  float* out = (float*)d_out;

  char* ws = (char*)d_ws;
  size_t off = 0;
  auto alloc = [&](size_t bytes) -> void* {
    void* p = ws + off; off += (bytes + 255) & ~(size_t)255; return p;
  };
  _Float16* qn  = (_Float16*)alloc((size_t)NROWS * DM * 2);
  _Float16* kn  = (_Float16*)alloc((size_t)NROWS * DM * 2);
  _Float16* vn  = (_Float16*)alloc((size_t)NROWS * DM * 2);
  _Float16* wqh = (_Float16*)alloc((size_t)DM * DM * 2);
  _Float16* wkh = (_Float16*)alloc((size_t)DM * DM * 2);
  _Float16* wvh = (_Float16*)alloc((size_t)DM * DM * 2);
  _Float16* woh = (_Float16*)alloc((size_t)DM * DM * 2);
  _Float16* winh= (_Float16*)alloc((size_t)WROWS * DK * 2);
  _Float16* qp  = (_Float16*)alloc((size_t)NROWS * DM * 2);
  _Float16* kp  = (_Float16*)alloc((size_t)NROWS * DM * 2);
  _Float16* vT  = (_Float16*)alloc((size_t)BATCH * NH * DK * 1024 * 2);
  _Float16* ao  = (_Float16*)alloc((size_t)NROWS * DM * 2);

  const int nW = DM * DM;
  cvt_f16_kernel<<<(nW + 255) / 256, 256, 0, stream>>>(wq, wqh, nW);
  cvt_f16_kernel<<<(nW + 255) / 256, 256, 0, stream>>>(wk, wkh, nW);
  cvt_f16_kernel<<<(nW + 255) / 256, 256, 0, stream>>>(wv, wvh, nW);
  cvt_f16_kernel<<<(nW + 255) / 256, 256, 0, stream>>>(wo, woh, nW);
  const int nWin = WROWS * DK;
  cvt_f16_kernel<<<(nWin + 255) / 256, 256, 0, stream>>>(rel, winh, nWin);

  ln_f16_kernel<<<NROWS, 256, 0, stream>>>(q, ln_g, ln_b, qn);
  ln_f16_kernel<<<NROWS, 256, 0, stream>>>(k, ln_g, ln_b, kn);
  ln_f16_kernel<<<NROWS, 256, 0, stream>>>(v, ln_g, ln_b, vn);

  dim3 gg(NROWS / 32, DM / 256);   // 125 x 4 blocks, 4 waves/block, 64 cols/wave
  gemm_wmma_kernel<<<gg, 128, 0, stream>>>(qn, wqh, bq, nullptr, qp, nullptr, nullptr, 0);
  gemm_wmma_kernel<<<gg, 128, 0, stream>>>(kn, wkh, bk, nullptr, kp, nullptr, nullptr, 0);
  gemm_wmma_kernel<<<gg, 128, 0, stream>>>(vn, wvh, bv, nullptr, nullptr, vT, nullptr, 1);

  attn_kernel<<<BATCH * NH * NIB, 32, 0, stream>>>(qp, kp, vT, winh, ao);

  gemm_wmma_kernel<<<gg, 128, 0, stream>>>(ao, woh, bo, q, nullptr, nullptr, out, 2);
}
